// Conv2d_mvm_63230508531833
// MI455X (gfx1250) — matrix operand order repaired and checked by reading, not compiled, not run on hardware
//
#include <hip/hip_runtime.h>

// ---------------------------------------------------------------------------
// Conv2d 3x3 s1 p1, Cin=Cout=16, 1024x1024, fp32 in/out.
// Implicit GEMM on CDNA5 WMMA: D(16px x 16co) = A(16px x K) * B(K x 16co),
// K = 144 = Cin*9 taps, chunked K=32 for v_wmma_f32_16x16x32_bf16.
// fp32 emulated via bf16 hi/lo split: Xh*Wh + Xl*Wh + Xh*Wl (f32 accum).
// The split is done ONCE at staging time; LDS holds X already in the packed
// bf16-pair WMMA-A element order so each A-fragment half is one ds_load_b128.
// ---------------------------------------------------------------------------

typedef __attribute__((ext_vector_type(16))) __bf16 v16bf;
typedef __attribute__((ext_vector_type(8)))  float  v8f;

static constexpr int CIN   = 16;
static constexpr int IMG_H = 1024;
static constexpr int IMG_W = 1024;
static constexpr int HWSZ  = IMG_H * IMG_W;
static constexpr int TILE_X  = 128;        // pixels per block (8 waves x 16 px)
static constexpr int WAVE_PX = 16;
static constexpr int NCHUNK  = 5;          // ceil(144 / 32)
static constexpr int TILE_W  = TILE_X + 2; // 130: halo of 1 each side
static constexpr int XPAIRS  = 8;          // 16 channels as 8 packed bf16 pairs

__device__ __forceinline__ unsigned short f2bf(float f) {
    // round-to-nearest-even fp32 -> bf16
    unsigned u = __float_as_uint(f);
    u += 0x7FFFu + ((u >> 16) & 1u);
    return (unsigned short)(u >> 16);
}
__device__ __forceinline__ float bf2f(unsigned short h) {
    return __uint_as_float(((unsigned)h) << 16);
}
__device__ __forceinline__ void split2(float a, float b, unsigned& hi, unsigned& lo) {
    const unsigned short h0 = f2bf(a), h1 = f2bf(b);
    const unsigned short l0 = f2bf(a - bf2f(h0));
    const unsigned short l1 = f2bf(b - bf2f(h1));
    hi = (unsigned)h0 | ((unsigned)h1 << 16);
    lo = (unsigned)l0 | ((unsigned)l1 << 16);
}

union Frag { unsigned u[8]; v16bf v; };

__global__ __launch_bounds__(256)
void conv2d_mvm_wmma(const float* __restrict__ x,
                     const float* __restrict__ wgt,
                     float* __restrict__ out) {
    // X tile, pre-split: [3 rows][130 x][8 packed ci-pairs] of bf16x2.
    __shared__ __align__(16) unsigned sXhi[3 * TILE_W * XPAIRS];
    __shared__ __align__(16) unsigned sXlo[3 * TILE_W * XPAIRS];
    // Weight fragments in WMMA-B lane layout, per K-chunk, hi/lo bf16 pairs.
    __shared__ __align__(32) unsigned sWhi[NCHUNK * 32 * 8];
    __shared__ __align__(32) unsigned sWlo[NCHUNK * 32 * 8];

    const int tid = threadIdx.x;
    const int y   = blockIdx.y;            // output row
    const int x0  = blockIdx.x * TILE_X;   // first output pixel of block

    // ---- stage input tile: load fp32 pair, split hi/lo once, pack ------
    for (int idx = tid; idx < 3 * TILE_W * XPAIRS; idx += 256) {
        const int ci = (idx & (XPAIRS - 1)) * 2;
        const int xl = (idx / XPAIRS) % TILE_W;
        const int r  = idx / (TILE_W * XPAIRS);
        const int gy = y + r - 1;
        const int gx = x0 + xl - 1;
        float a = 0.0f, b = 0.0f;
        if ((unsigned)gy < (unsigned)IMG_H && (unsigned)gx < (unsigned)IMG_W) {
            const float* p = x + gy * IMG_W + gx;
            a = p[ci * HWSZ];
            b = p[(ci + 1) * HWSZ];
        }
        unsigned hi, lo;
        split2(a, b, hi, lo);
        sXhi[idx] = hi;
        sXlo[idx] = lo;
    }

    // ---- pack weights into per-chunk WMMA-B fragments (hi/lo) ----------
    // B layout (32x16, K x cout): lanes 0-15 hold K=0..7,16..23, lanes 16-31
    // hold K=8..15,24..31 (V j<4: 8h+2j, j>=4: 16+8h+2(j-4)); column n = lane & 15.
    for (int idx = tid; idx < NCHUNK * 32 * 8; idx += 256) {
        const int j    = idx & 7;
        const int lane = (idx >> 3) & 31;
        const int c    = idx >> 8;
        const int n    = lane & 15;                    // cout column
        const int klo  = ((j >> 2) << 4) + ((lane >> 4) << 3) + 2 * (j & 3);   // even local-K
        const int tap  = 2 * c + (klo >> 4);           // 0..9 (9 => zero pad)
        unsigned hi = 0u, lo = 0u;
        if (tap <= 8) {
            const int ky  = tap / 3, kx = tap % 3;
            const int ci0 = klo & 15;
            const float w0 = wgt[((n * CIN + ci0    ) * 3 + ky) * 3 + kx];
            const float w1 = wgt[((n * CIN + ci0 + 1) * 3 + ky) * 3 + kx];
            split2(w0, w1, hi, lo);
        }
        sWhi[idx] = hi;
        sWlo[idx] = lo;
    }

    __syncthreads();

    // ---- per-wave 16px x 16cout WMMA tile ------------------------------
    const int wave  = tid >> 5;
    const int lane  = tid & 31;
    const int p     = lane & 15;           // pixel index within wave tile
    const int cioff = (lane >> 4) << 2;    // A: lanes 16-31 hold odd K-octets
    const int px    = wave * WAVE_PX + p;  // pixel within block tile

    v8f acc = {0.f, 0.f, 0.f, 0.f, 0.f, 0.f, 0.f, 0.f};

#pragma unroll
    for (int c = 0; c < NCHUNK; ++c) {
        Frag ahi, alo;
#pragma unroll
        for (int hh = 0; hh < 2; ++hh) {
            const int tap = 2 * c + hh;
            if (tap <= 8) {
                const int ky = tap / 3, kx = tap % 3;
                const int base = ((ky * TILE_W) + (px + kx)) * XPAIRS + cioff;
                const uint4 h4 = *reinterpret_cast<const uint4*>(&sXhi[base]);
                const uint4 l4 = *reinterpret_cast<const uint4*>(&sXlo[base]);
                ahi.u[4 * hh + 0] = h4.x; ahi.u[4 * hh + 1] = h4.y;
                ahi.u[4 * hh + 2] = h4.z; ahi.u[4 * hh + 3] = h4.w;
                alo.u[4 * hh + 0] = l4.x; alo.u[4 * hh + 1] = l4.y;
                alo.u[4 * hh + 2] = l4.z; alo.u[4 * hh + 3] = l4.w;
            } else {
#pragma unroll
                for (int j = 0; j < 4; ++j) {
                    ahi.u[4 * hh + j] = 0u;
                    alo.u[4 * hh + j] = 0u;
                }
            }
        }
        const v16bf bhi = *reinterpret_cast<const v16bf*>(&sWhi[(c * 32 + lane) * 8]);
        const v16bf blo = *reinterpret_cast<const v16bf*>(&sWlo[(c * 32 + lane) * 8]);
        // 3-product bf16 split ~ fp32: Xh*Wh + Xl*Wh + Xh*Wl
        acc = __builtin_amdgcn_wmma_f32_16x16x32_bf16(false, ahi.v, false, bhi,
                                                      (short)0, acc, false, false);
        acc = __builtin_amdgcn_wmma_f32_16x16x32_bf16(false, alo.v, false, bhi,
                                                      (short)0, acc, false, false);
        acc = __builtin_amdgcn_wmma_f32_16x16x32_bf16(false, ahi.v, false, blo,
                                                      (short)0, acc, false, false);
    }

    // ---- store D: lanes 0-15 -> M=j (N=lane); lanes 16-31 -> M=j+8 -----
    const int n     = lane & 15;                   // cout
    const int mbase = (lane >> 4) << 3;
    float* op = out + n * HWSZ + y * IMG_W + x0 + wave * WAVE_PX + mbase;
    float4 lo4, hi4;
    lo4.x = acc[0]; lo4.y = acc[1]; lo4.z = acc[2]; lo4.w = acc[3];
    hi4.x = acc[4]; hi4.y = acc[5]; hi4.z = acc[6]; hi4.w = acc[7];
    *reinterpret_cast<float4*>(op)     = lo4;
    *reinterpret_cast<float4*>(op + 4) = hi4;
}

extern "C" void kernel_launch(void* const* d_in, const int* in_sizes, int n_in,
                              void* d_out, int out_size, void* d_ws, size_t ws_size,
                              hipStream_t stream) {
    const float* x = (const float*)d_in[0];   // (1,16,1024,1024) fp32
    const float* w = (const float*)d_in[1];   // (16,16,3,3) fp32
    float* out     = (float*)d_out;           // (16,1024,1024) fp32
    (void)in_sizes; (void)n_in; (void)out_size; (void)d_ws; (void)ws_size;

    dim3 grid(IMG_W / TILE_X, IMG_H);         // 8 x 1024 blocks
    conv2d_mvm_wmma<<<grid, 256, 0, stream>>>(x, w, out);
}
